// ODEBlock_45775761441050
// MI455X (gfx1250) — matrix operand order repaired and checked by reading, not compiled, not run on hardware
//
#include <hip/hip_runtime.h>
#include <hip/hip_bf16.h>

typedef _Float16 v16h __attribute__((ext_vector_type(16)));
typedef _Float16 v8h  __attribute__((ext_vector_type(8)));
typedef _Float16 v4h  __attribute__((ext_vector_type(4)));
typedef float    v8f  __attribute__((ext_vector_type(8)));

#define NELEM   (64*64*32*32)   // 4,194,304
#define SXW     72              // padded channel stride in LDS (halves)
#define SXROWS  (6*34)          // 6 h-rows x (32 w + 2 halo)

// ---------------------------------------------------------------------------
// Weight prepack: conv_w [64][65][3][3] f32  ->  wpk [tap r][co][ci] f16
// (ci excludes the t-channel; channel 0 tap weights go to t9[co*9+r] f32)
// ---------------------------------------------------------------------------
__global__ void prepack_k(const float* __restrict__ w,
                          _Float16* __restrict__ wpk,
                          float* __restrict__ t9)
{
    int i = blockIdx.x * 256 + threadIdx.x;          // grid covers 36864
    if (i < 64*64*9) {
        int co = i / 576;
        int rem = i - co * 576;
        int ci = rem / 9;
        int r  = rem - ci * 9;
        wpk[(r*64 + co)*64 + ci] = (_Float16)w[(co*65 + ci + 1)*9 + r];
    }
    if (i < 576) {
        int co = i / 9, r = i - co * 9;
        t9[i] = w[(co*65)*9 + r];
    }
}

// ---------------------------------------------------------------------------
// y = x (f32 state), xh = f16(x) (conv input for k1)
// ---------------------------------------------------------------------------
__global__ void init_k(const float* __restrict__ x,
                       float* __restrict__ y,
                       _Float16* __restrict__ xh)
{
    int i = blockIdx.x * 256 + threadIdx.x;
    float v = x[i];
    y[i]  = v;
    xh[i] = (_Float16)v;
}

// ---------------------------------------------------------------------------
// conv3x3 (65->64, pad 1) as implicit GEMM with v_wmma_f32_16x16x32_f16.
//   xin : f16 NCHW [64][64][32][32]   (the 64 "real" channels)
//   wpk : f16 [9][64 co][64 ci]
//   t9  : f32 [64 co][9]  weights for the constant-t channel
// Output f32 NCHW. Grid = 512 (b * h/4), block = 256 (8 waves).
// ---------------------------------------------------------------------------
__launch_bounds__(256)
__global__ void conv3x3_wmma(const _Float16* __restrict__ xin,
                             const _Float16* __restrict__ wpk,
                             const float* __restrict__ t9,
                             const float* __restrict__ bias,
                             float t,
                             float* __restrict__ out)
{
    __shared__ __align__(16) _Float16 sx[SXROWS * SXW];
    __shared__ float t9s[576];
    __shared__ float bs[64];

    const int tid = threadIdx.x;
    const int blk = blockIdx.x;
    const int b   = blk >> 3;
    const int h0  = (blk & 7) << 2;

    // zero LDS staging (covers OOB rows and w-halo uniformly)
    {
        unsigned long long* z = (unsigned long long*)sx;
        for (int i = tid; i < (SXROWS * SXW) / 4; i += 256) z[i] = 0ULL;
    }
    for (int i = tid; i < 576; i += 256) t9s[i] = t9[i];
    if (tid < 64) bs[tid] = bias[tid];
    __syncthreads();

    // stage input rows h0-1 .. h0+4, channel-interleaved f16
    {
        const int w  = tid & 31;
        const int cg = tid >> 5;                      // 0..7
        for (int row = 0; row < 6; ++row) {
            int hh = h0 - 1 + row;
            if (hh < 0 || hh > 31) continue;
            for (int cc = 0; cc < 2; ++cc) {
                int ci0 = (cg * 2 + cc) * 4;          // 0,4,...,60
                const _Float16* gp = xin + (b*64 + ci0)*1024 + hh*32 + w;
                v4h v;
                v[0] = gp[0];
                v[1] = gp[1024];
                v[2] = gp[2048];
                v[3] = gp[3072];
                *(v4h*)&sx[(row*34 + 1 + w)*SXW + ci0] = v;
            }
        }
    }
    __syncthreads();

    const int wv    = tid >> 5;
    const int lane  = tid & 31;
    const int ln    = lane & 15;
    const int hi    = lane >> 4;
    const int mpair = wv & 1;    // co half: 0 -> co 0..31, 1 -> co 32..63
    const int nq    = wv >> 1;   // which of the 4 output rows

    v8f zero = {0.f,0.f,0.f,0.f,0.f,0.f,0.f,0.f};
    v8f acc[2][2];
    acc[0][0] = zero; acc[0][1] = zero; acc[1][0] = zero; acc[1][1] = zero;

    for (int r = 0; r < 9; ++r) {                 // tap loop
        const int dh = r / 3 - 1;
        const int dw = r - (r / 3) * 3 - 1;
        const int srow = nq + 1 + dh;             // 0..5
#pragma unroll
        for (int cb = 0; cb < 2; ++cb) {          // 32-channel K chunks
            const int c0 = cb * 32;
            v16h af[2], bf[2];
#pragma unroll
            for (int f = 0; f < 2; ++f) {
                const _Float16* ap =
                    wpk + ((r*64 + mpair*32 + f*16 + ln)*64 + c0 + 8*hi);
                v8h a0 = *(const v8h*)ap;
                v8h a1 = *(const v8h*)(ap + 16);
                af[f] = __builtin_shufflevector(a0, a1,
                        0,1,2,3,4,5,6,7,8,9,10,11,12,13,14,15);
            }
#pragma unroll
            for (int g = 0; g < 2; ++g) {
                const _Float16* bp =
                    &sx[(srow*34 + 1 + dw + g*16 + ln)*SXW + c0 + 8*hi];
                v8h b0 = *(const v8h*)bp;
                v8h b1 = *(const v8h*)(bp + 16);
                bf[g] = __builtin_shufflevector(b0, b1,
                        0,1,2,3,4,5,6,7,8,9,10,11,12,13,14,15);
            }
#pragma unroll
            for (int f = 0; f < 2; ++f)
#pragma unroll
                for (int g = 0; g < 2; ++g)
                    acc[f][g] = __builtin_amdgcn_wmma_f32_16x16x32_f16(
                        false, af[f], false, bf[g],
                        (short)0, acc[f][g], false, false);
        }
    }

    // ------------------------------------------------------------------
    // Branch-free epilogue: bias + t-channel contribution, store f32.
    //   h-masks (kh=0/2 validity) are wave-uniform; w-masks only differ
    //   at global w==0 (left tap) and w==31 (right tap).
    // ------------------------------------------------------------------
    const int   h   = h0 + nq;
    const float mh0 = (h >= 1)  ? 1.f : 0.f;   // kh=0 row valid
    const float mh2 = (h <= 30) ? 1.f : 0.f;   // kh=2 row valid
    const int   w0_ = ln;                      // g=0 column
    const int   w1_ = 16 + ln;                 // g=1 column
    const float a0  = (w0_ >= 1)  ? 1.f : 0.f; // left tap valid, g=0 (g=1 always 1)
    const float b1  = (w1_ <= 30) ? 1.f : 0.f; // right tap valid, g=1 (g=0 always 1)

    float* orow = out + (size_t)(b*64) * 1024 + h*32;
#pragma unroll
    for (int f = 0; f < 2; ++f) {
#pragma unroll
        for (int i = 0; i < 8; ++i) {
            const int co = mpair*32 + f*16 + i + 8*hi;
            const float* tp = &t9s[co*9];
            const float t0 = tp[0], t1 = tp[1], t2 = tp[2];
            const float t3 = tp[3], t4 = tp[4], t5 = tp[5];
            const float t6 = tp[6], t7 = tp[7], t8 = tp[8];
            // g=0: mw_left = a0, mw_right = 1
            const float tc0 = mh0 * (a0*t0 + t1 + t2)
                            +       (a0*t3 + t4 + t5)
                            + mh2 * (a0*t6 + t7 + t8);
            // g=1: mw_left = 1, mw_right = b1
            const float tc1 = mh0 * (t0 + t1 + b1*t2)
                            +       (t3 + t4 + b1*t5)
                            + mh2 * (t6 + t7 + b1*t8);
            const float base = bs[co];
            orow[(size_t)co*1024 + w0_] = acc[f][0][i] + base + t*tc0;
            orow[(size_t)co*1024 + w1_] = acc[f][1][i] + base + t*tc1;
        }
    }
}

// ---------------------------------------------------------------------------
// GroupNorm(32 groups of 2 channels) + optional ReLU, f32 in -> f16 out.
// Grid = 64*32 blocks, 256 threads; each block owns 2 ch x 1024 px.
// ---------------------------------------------------------------------------
__global__ void groupnorm_k(const float* __restrict__ in,
                            const float* __restrict__ sc,
                            const float* __restrict__ bi,
                            _Float16* __restrict__ out, int relu)
{
    __shared__ float sa[256], sb[256];
    const int tid = threadIdx.x;
    const int b = blockIdx.x >> 5;
    const int g = blockIdx.x & 31;
    const float* p = in + (size_t)(b*64 + g*2) * 1024;

    float v[8];
    float s = 0.f, q = 0.f;
#pragma unroll
    for (int j = 0; j < 8; ++j) {
        v[j] = p[tid + 256*j];
        s += v[j];
        q += v[j] * v[j];
    }
    sa[tid] = s; sb[tid] = q;
    __syncthreads();
    for (int off = 128; off > 0; off >>= 1) {
        if (tid < off) { sa[tid] += sa[tid+off]; sb[tid] += sb[tid+off]; }
        __syncthreads();
    }
    const float mu  = sa[0] * (1.f/2048.f);
    const float inv = rsqrtf(sb[0] * (1.f/2048.f) - mu*mu + 1e-5f);

    const float s0 = sc[g*2],   s1 = sc[g*2+1];
    const float b0 = bi[g*2],   b1 = bi[g*2+1];
    _Float16* o = out + (size_t)(b*64 + g*2) * 1024;
#pragma unroll
    for (int j = 0; j < 8; ++j) {
        const int c = j >> 2;   // j<4 -> channel g*2, else g*2+1
        float r = (v[j] - mu) * inv * (c ? s1 : s0) + (c ? b1 : b0);
        if (relu) r = fmaxf(r, 0.f);
        o[tid + 256*j] = (_Float16)r;
    }
}

// ---------------------------------------------------------------------------
// RK combine: v = y + sum_{j<nk} c_j * k_j ; xh = f16(v); optional f32 out.
// ---------------------------------------------------------------------------
__global__ void combine_k(const float* __restrict__ y,
                          const _Float16* __restrict__ k0,
                          const _Float16* __restrict__ k1,
                          const _Float16* __restrict__ k2,
                          const _Float16* __restrict__ k3,
                          const _Float16* __restrict__ k4,
                          const _Float16* __restrict__ k5,
                          float c0, float c1, float c2,
                          float c3, float c4, float c5,
                          int nk,
                          _Float16* __restrict__ xo,
                          float* __restrict__ yo)
{
    int i = blockIdx.x * 256 + threadIdx.x;
    float v = y[i];
    if (nk > 0) v += c0 * (float)k0[i];
    if (nk > 1) v += c1 * (float)k1[i];
    if (nk > 2) v += c2 * (float)k2[i];
    if (nk > 3) v += c3 * (float)k3[i];
    if (nk > 4) v += c4 * (float)k4[i];
    if (nk > 5) v += c5 * (float)k5[i];
    xo[i] = (_Float16)v;
    if (yo) yo[i] = v;
}

// ---------------------------------------------------------------------------
// Host orchestration
// ---------------------------------------------------------------------------
extern "C" void kernel_launch(void* const* d_in, const int* in_sizes, int n_in,
                              void* d_out, int out_size, void* d_ws, size_t ws_size,
                              hipStream_t stream)
{
    (void)in_sizes; (void)n_in; (void)out_size; (void)ws_size;
    const float* x   = (const float*)d_in[0];
    const float* c1w = (const float*)d_in[1];
    const float* c1b = (const float*)d_in[2];
    const float* g1s = (const float*)d_in[3];
    const float* g1b = (const float*)d_in[4];
    const float* c2w = (const float*)d_in[5];
    const float* c2b = (const float*)d_in[6];
    const float* g2s = (const float*)d_in[7];
    const float* g2b = (const float*)d_in[8];

    size_t off = 0;
    auto carve = [&](size_t bytes) -> void* {
        void* r = (char*)d_ws + off;
        off = (off + bytes + 255) & ~(size_t)255;
        return r;
    };
    float*     y     = (float*)    carve((size_t)NELEM * 4);
    _Float16*  xh    = (_Float16*) carve((size_t)NELEM * 2);
    float*     convb = (float*)    carve((size_t)NELEM * 4);
    _Float16*  h1h   = (_Float16*) carve((size_t)NELEM * 2);
    _Float16*  kb[6];
    for (int i = 0; i < 6; ++i) kb[i] = (_Float16*)carve((size_t)NELEM * 2);
    _Float16*  wpk1  = (_Float16*) carve(9*64*64*2);
    _Float16*  wpk2  = (_Float16*) carve(9*64*64*2);
    float*     t9a   = (float*)    carve(576*4);
    float*     t9b   = (float*)    carve(576*4);

    prepack_k<<<144, 256, 0, stream>>>(c1w, wpk1, t9a);
    prepack_k<<<144, 256, 0, stream>>>(c2w, wpk2, t9b);
    init_k<<<NELEM/256, 256, 0, stream>>>(x, y, xh);

    static const float Cc[5] = {0.161f, 0.327f, 0.9f, 0.9800255409045097f, 1.0f};
    static const float A1[1] = {0.161f};
    static const float A2[2] = {-0.008480655492356989f, 0.335480655492357f};
    static const float A3[3] = {2.8971530571054935f, -6.359448489975075f, 4.3622954328695815f};
    static const float A4[4] = {5.325864828439257f, -11.748883564062828f,
                                7.4955393428898365f, -0.09249506636175525f};
    static const float A5[5] = {5.86145544294642f, -12.92096931784711f, 8.159367898576159f,
                                -0.071584973281401f, -0.028269050394068383f};
    static const float Bc[6] = {0.09646076681806523f, 0.01f, 0.4798896504144996f,
                                1.379008574103742f, -3.290069515436081f, 2.324710524099774f};

    const float dt = 0.125f;
    float t = 0.f;

    auto ode = [&](float tt, const _Float16* inh, _Float16* kout) {
        conv3x3_wmma<<<512, 256, 0, stream>>>(inh, wpk1, t9a, c1b, tt, convb);
        groupnorm_k<<<2048, 256, 0, stream>>>(convb, g1s, g1b, h1h, 1);
        conv3x3_wmma<<<512, 256, 0, stream>>>(h1h, wpk2, t9b, c2b, tt, convb);
        groupnorm_k<<<2048, 256, 0, stream>>>(convb, g2s, g2b, kout, 0);
    };
    auto comb = [&](const float* a, int nk, float* yo) {
        float c[6] = {0,0,0,0,0,0};
        for (int j = 0; j < nk; ++j) c[j] = dt * a[j];
        combine_k<<<NELEM/256, 256, 0, stream>>>(y, kb[0], kb[1], kb[2], kb[3], kb[4], kb[5],
                                                 c[0], c[1], c[2], c[3], c[4], c[5],
                                                 nk, xh, yo);
    };

    for (int s = 0; s < 8; ++s) {
        ode(t, xh, kb[0]);
        comb(A1, 1, nullptr); ode(t + Cc[0]*dt, xh, kb[1]);
        comb(A2, 2, nullptr); ode(t + Cc[1]*dt, xh, kb[2]);
        comb(A3, 3, nullptr); ode(t + Cc[2]*dt, xh, kb[3]);
        comb(A4, 4, nullptr); ode(t + Cc[3]*dt, xh, kb[4]);
        comb(A5, 5, nullptr); ode(t + Cc[4]*dt, xh, kb[5]);
        comb(Bc, 6, (s == 7) ? (float*)d_out : y);
        t += dt;
    }
}
